// GraphFlowFeatureExtractor_48361331753074
// MI455X (gfx1250) — matrix operand order repaired and checked by reading, not compiled, not run on hardware
//
#include <hip/hip_runtime.h>
#include <hip/hip_bf16.h>

// ---------------- constants from the reference ----------------
#define C_N_FEAT 32
#define C_D 16
#define C_CTX 32
#define C_HID 256
#define C_T 4
#define C_K 8
#define C_BT 4096
#define C_OUTW (C_D * (3 * C_K - 1))   // 368
#define C_OUTP 384                      // padded to multiple of 64
#define C_IN0 (C_D + C_CTX)             // 48
#define C_IN0P 64                       // padded K for layer 0
#define C_BOUND 5.0f
#define HALF_LOG_2PI 0.91893853320467274178f

typedef __attribute__((ext_vector_type(16))) __bf16 v16bf;
typedef __attribute__((ext_vector_type(8)))  __bf16 v8bf;
typedef __attribute__((ext_vector_type(8)))  float  v8f;

// ---------------- helpers ----------------
__device__ __forceinline__ float eluf(float v) { return v > 0.f ? v : expm1f(v); }
__device__ __forceinline__ int deg_h(int j) { return (j % (C_D - 1)) + 1; }
__device__ __forceinline__ int rank_of(int d, int parity) { return parity ? (C_D - 1 - d) : d; }

// MADE mask (i = input row of W, j = output col). 0=none,1=m0,2=mh,3=mo
__device__ __forceinline__ float mask_val(int mode, int parity, int i, int j) {
  if (mode == 0) return 1.f;
  if (mode == 1) {
    int di = (i < C_D) ? (rank_of(i, parity) + 1) : 0;
    return (deg_h(j) >= di) ? 1.f : 0.f;
  }
  if (mode == 2) return (deg_h(j) >= deg_h(i)) ? 1.f : 0.f;
  int dout = rank_of(j / (3 * C_K - 1), parity) + 1;
  return (dout > deg_h(i)) ? 1.f : 0.f;
}

// ---------------- weight prep: transpose + mask + bf16, zero-padded ----------------
// W[K,N] fp32  ->  Wt[Npad][Kpad] bf16
__global__ void prep_wt(const float* __restrict__ W, __bf16* __restrict__ Wt,
                        int K, int N, int Kpad, int Npad, int mode, int parity) {
  int tid = blockIdx.x * blockDim.x + threadIdx.x;
  int total = Npad * Kpad;
  if (tid >= total) return;
  int c = tid / Kpad, k = tid % Kpad;
  float v = 0.f;
  if (k < K && c < N) v = W[(size_t)k * N + c] * mask_val(mode, parity, k, c);
  Wt[tid] = (__bf16)v;
}

__global__ void f32_to_bf16(const float* __restrict__ in, __bf16* __restrict__ out,
                            long long n) {
  long long tid = (long long)blockIdx.x * blockDim.x + threadIdx.x;
  if (tid < n) out[tid] = (__bf16)in[tid];
}

// ---------------- WMMA fragment loaders (ISA 7.12.2, wave32) ----------------
// A 16x32 bf16: lane's 16 elems = A[r][k0+8*hi .. +8) ++ A[r][k0+16+8*hi .. +8)
__device__ __forceinline__ v16bf load_a_frag(const __bf16* __restrict__ A,
                                             size_t rowbase, int k0, int hi) {
  const v8bf* p = (const v8bf*)(A + rowbase + k0 + 8 * hi);
  v8bf c0 = p[0];
  v8bf c1 = p[2];
  return __builtin_shufflevector(c0, c1, 0, 1, 2, 3, 4, 5, 6, 7,
                                 8, 9, 10, 11, 12, 13, 14, 15);
}

// B 32x16 bf16 from transposed Wt[Npad][Kpad]: lane's 16 elems = Wt[c][k0+8*hi .. +8) ++ Wt[c][k0+16+8*hi .. +8)
__device__ __forceinline__ v16bf load_b_frag(const __bf16* __restrict__ Wt,
                                             size_t colbase, int k0, int hi) {
  const v8bf* p = (const v8bf*)(Wt + colbase + k0 + 8 * hi);
  v8bf b0 = p[0], b1 = p[2];
  return __builtin_shufflevector(b0, b1, 0, 1, 2, 3, 4, 5, 6, 7,
                                 8, 9, 10, 11, 12, 13, 14, 15);
}

// ---------------- bf16 WMMA GEMM ----------------
// C[M,N] = act( A[M,Kpad]bf16 @ Wt^T + (A2 ? A2 @ Wt2^T : 0) + bias )
// grid: (ceil(M/32), Npad/64), block 128 (4 waves; each wave: 32x16 tile, 2 accums)
// Software-pipelined; row index clamped (no exec-masked loads); final prefetch
// overreads <=64B past logical end (workspace allocs are padded).
__global__ void gemm_bf16(const __bf16* __restrict__ A, int lda,
                          const __bf16* __restrict__ A2, int lda2,
                          const __bf16* __restrict__ Wt, const __bf16* __restrict__ Wt2, int ldb,
                          const float* __restrict__ bias,
                          float* __restrict__ Cf, __bf16* __restrict__ Cb, int ldc,
                          int M, int N, int K, int K2, int do_elu) {
  int lane = threadIdx.x & 31;
  int wave = threadIdx.x >> 5;
  int row0 = blockIdx.x * 32;
  int col0 = (blockIdx.y * 4 + wave) * 16;
  int hi = lane >> 4;
  int lo16 = lane & 15;
  int r0 = min(row0 + lo16, M - 1);
  int r1 = min(row0 + 16 + lo16, M - 1);
  int c = col0 + lo16;
  size_t cb = (size_t)c * ldb;

  v8f acc0 = {}, acc1 = {};
  {
    size_t rb0 = (size_t)r0 * lda, rb1 = (size_t)r1 * lda;
    v16bf a0 = load_a_frag(A, rb0, 0, hi);
    v16bf a1 = load_a_frag(A, rb1, 0, hi);
    v16bf b  = load_b_frag(Wt, cb, 0, hi);
    for (int k0 = 32; k0 <= K; k0 += 32) {
      v16bf na0 = load_a_frag(A, rb0, k0, hi);
      v16bf na1 = load_a_frag(A, rb1, k0, hi);
      v16bf nb  = load_b_frag(Wt, cb, k0, hi);
      acc0 = __builtin_amdgcn_wmma_f32_16x16x32_bf16(false, a0, false, b, (short)0, acc0, false, false);
      acc1 = __builtin_amdgcn_wmma_f32_16x16x32_bf16(false, a1, false, b, (short)0, acc1, false, false);
      a0 = na0; a1 = na1; b = nb;
    }
  }
  if (A2) {
    size_t rb0 = (size_t)r0 * lda2, rb1 = (size_t)r1 * lda2;
    v16bf a0 = load_a_frag(A2, rb0, 0, hi);
    v16bf a1 = load_a_frag(A2, rb1, 0, hi);
    v16bf b  = load_b_frag(Wt2, cb, 0, hi);
    for (int k0 = 32; k0 <= K2; k0 += 32) {
      v16bf na0 = load_a_frag(A2, rb0, k0, hi);
      v16bf na1 = load_a_frag(A2, rb1, k0, hi);
      v16bf nb  = load_b_frag(Wt2, cb, k0, hi);
      acc0 = __builtin_amdgcn_wmma_f32_16x16x32_bf16(false, a0, false, b, (short)0, acc0, false, false);
      acc1 = __builtin_amdgcn_wmma_f32_16x16x32_bf16(false, a1, false, b, (short)0, acc1, false, false);
      a0 = na0; a1 = na1; b = nb;
    }
  }

  bool cok = c < N;
  float bv = cok ? bias[c] : 0.f;
#pragma unroll
  for (int i = 0; i < 8; ++i) {
    int rr0 = row0 + i + 8 * hi;
    int rr1 = rr0 + 16;
    if (rr0 < M && cok) {
      float v = acc0[i] + bv;
      if (do_elu) v = eluf(v);
      if (Cf) Cf[(size_t)rr0 * ldc + c] = v;
      else    Cb[(size_t)rr0 * ldc + c] = (__bf16)v;
    }
    if (rr1 < M && cok) {
      float v = acc1[i] + bv;
      if (do_elu) v = eluf(v);
      if (Cf) Cf[(size_t)rr1 * ldc + c] = v;
      else    Cb[(size_t)rr1 * ldc + c] = (__bf16)v;
    }
  }
}

// ---------------- graph aggregation ----------------
__global__ void scatter_accum(const __bf16* __restrict__ feat, const int* __restrict__ src,
                              const int* __restrict__ dst, float* __restrict__ accum,
                              float* cnt, int E, int F) {
  int per = F >> 3;
  long long tid = (long long)blockIdx.x * blockDim.x + threadIdx.x;
  long long total = (long long)E * per;
  if (tid >= total) return;
  int e = (int)(tid / per), part = (int)(tid % per);
  int s = src[e], d = dst[e];
  v8bf v = *(const v8bf*)(feat + (long long)s * F + part * 8);
  float* acc = accum + (long long)d * F + part * 8;
#pragma unroll
  for (int i = 0; i < 8; ++i) atomicAdd(acc + i, (float)v[i]);
  if (cnt && part == 0) atomicAdd(cnt + d, 1.0f);
}

// accum fp32 / cnt -> bf16 mean
__global__ void div_cnt_bf16(const float* __restrict__ accum, const float* __restrict__ cnt,
                             __bf16* __restrict__ out, long long total, int F) {
  long long tid = (long long)blockIdx.x * blockDim.x + threadIdx.x;
  if (tid >= total) return;
  out[tid] = (__bf16)(accum[tid] / fmaxf(cnt[tid / F], 1.0f));
}

// row L2-normalize then ELU; blockDim.x == F (64 or 128); writes bf16 or fp32(in place)
__global__ void rownorm_elu(const float* __restrict__ hin, __bf16* __restrict__ outb,
                            float* __restrict__ outf, int F) {
  __shared__ float red[128];
  size_t row = blockIdx.x;
  int j = threadIdx.x;
  float v = hin[row * F + j];
  red[j] = v * v;
  __syncthreads();
  for (int s2 = F >> 1; s2 > 0; s2 >>= 1) {
    if (j < s2) red[j] += red[j + s2];
    __syncthreads();
  }
  float o = eluf(v / fmaxf(sqrtf(red[0]), 1e-12f));
  if (outb) outb[row * F + j] = (__bf16)o;
  else      outf[row * F + j] = o;
}

// ---------------- global max pool + ctx ----------------
__device__ __forceinline__ void atomicMaxFloat(float* addr, float val) {
  unsigned* ua = (unsigned*)addr;
  unsigned old = *ua;
  while (true) {
    float f = __uint_as_float(old);
    if (f >= val) break;
    unsigned assumed = old;
    old = atomicCAS(ua, assumed, __float_as_uint(val));
    if (old == assumed) break;
  }
}

__global__ void init_gmax(float* g) { if (threadIdx.x < 64) g[threadIdx.x] = -1e30f; }

__global__ void colmax(const float* __restrict__ h, float* __restrict__ gmax, int N, int F) {
  int j = threadIdx.x;  // F==64 threads
  float m = -1e30f;
  for (int r = blockIdx.x; r < N; r += gridDim.x) m = fmaxf(m, h[(long long)r * F + j]);
  atomicMaxFloat(&gmax[j], m);
}

__global__ void ctx_proj(const float* __restrict__ gmax, const float* __restrict__ wfc,
                         const float* __restrict__ bfc, float* __restrict__ ctx) {
  int j = threadIdx.x;
  if (j >= C_CTX) return;
  float acc = bfc[j];
  for (int i = 0; i < 2 * C_N_FEAT; ++i) acc += eluf(gmax[i]) * wfc[i * C_CTX + j];
  ctx[j] = acc;
}

// ---------------- flow ----------------
// hdin[BT][64] bf16: [z(16) | ctx(32) | zeros(16)]
__global__ void pack_hdin(const float* __restrict__ z, const float* __restrict__ ctx,
                          __bf16* __restrict__ hdin) {
  int tid = blockIdx.x * blockDim.x + threadIdx.x;
  if (tid >= C_BT * C_IN0P) return;
  int s = tid / C_IN0P, e = tid % C_IN0P;
  float v = 0.f;
  if (e < C_D) v = z[s * C_D + e];
  else if (e < C_IN0) v = ctx[e - C_D];
  hdin[tid] = (__bf16)v;
}

// rational-quadratic spline, one thread per sample
__global__ void rqs_kernel(const float* __restrict__ p, float* __restrict__ z,
                           float* __restrict__ logdet) {
  int s = blockIdx.x * blockDim.x + threadIdx.x;
  if (s >= C_BT) return;
  float ldacc = 0.f;
  for (int d = 0; d < C_D; ++d) {
    const float* pp = p + (long long)s * C_OUTW + d * (3 * C_K - 1);
    float w[C_K], h[C_K], kx[C_K + 1], ky[C_K + 1], dv[C_K + 1];
    float mw = -1e30f, mh = -1e30f;
#pragma unroll
    for (int i = 0; i < C_K; ++i) {
      w[i] = pp[i];
      h[i] = pp[C_K + i];
      mw = fmaxf(mw, w[i]);
      mh = fmaxf(mh, h[i]);
    }
    float sw = 0.f, sh = 0.f;
#pragma unroll
    for (int i = 0; i < C_K; ++i) {
      w[i] = expf(w[i] - mw); sw += w[i];
      h[i] = expf(h[i] - mh); sh += h[i];
    }
    kx[0] = -C_BOUND; ky[0] = -C_BOUND;
#pragma unroll
    for (int i = 0; i < C_K; ++i) {
      kx[i + 1] = kx[i] + w[i] * (2.f * C_BOUND / sw);
      ky[i + 1] = ky[i] + h[i] * (2.f * C_BOUND / sh);
    }
    dv[0] = 1.f; dv[C_K] = 1.f;
#pragma unroll
    for (int i = 0; i < C_K - 1; ++i) {
      float dd = pp[2 * C_K + i];
      dv[i + 1] = (dd > 20.f) ? dd : log1pf(expf(dd));
    }
    float x = z[s * C_D + d];
    bool inside = (x > -C_BOUND) && (x < C_BOUND);
    float xc = fminf(fmaxf(x, -C_BOUND + 1e-6f), C_BOUND - 1e-6f);
    int idx = -1;
#pragma unroll
    for (int i = 0; i < C_K + 1; ++i) idx += (xc >= kx[i]) ? 1 : 0;
    idx = min(max(idx, 0), C_K - 1);
    float x0 = kx[idx], x1 = kx[idx + 1], y0 = ky[idx], y1 = ky[idx + 1];
    float d0 = dv[idx], d1 = dv[idx + 1];
    float sl = (y1 - y0) / (x1 - x0);
    float xi = (xc - x0) / (x1 - x0);
    float u = xi * (1.f - xi);
    float den = sl + (d0 + d1 - 2.f * sl) * u;
    float y = y0 + (y1 - y0) * (sl * xi * xi + d0 * u) / den;
    float ld = logf(sl * sl * (d1 * xi * xi + 2.f * sl * u + d0 * (1.f - xi) * (1.f - xi)))
             - 2.f * logf(den);
    z[s * C_D + d] = inside ? y : x;
    ldacc += inside ? ld : 0.f;
  }
  logdet[s] += ldacc;
}

__global__ void logp_kernel(const float* __restrict__ z, const float* __restrict__ logdet,
                            float* __restrict__ out) {
  int s = blockIdx.x * blockDim.x + threadIdx.x;
  if (s >= C_BT) return;
  float acc = 0.f;
#pragma unroll
  for (int d = 0; d < C_D; ++d) { float v = z[s * C_D + d]; acc -= 0.5f * v * v; }
  out[s] = acc - C_D * HALF_LOG_2PI + logdet[s];
}

// ---------------- orchestration ----------------
extern "C" void kernel_launch(void* const* d_in, const int* in_sizes, int n_in,
                              void* d_out, int out_size, void* d_ws, size_t ws_size,
                              hipStream_t stream) {
  (void)n_in; (void)out_size; (void)ws_size;
  const float* theta = (const float*)d_in[0];
  const float* x     = (const float*)d_in[1];
  const int*   ei    = (const int*)d_in[2];
  const float* w1l = (const float*)d_in[3];
  const float* w1r = (const float*)d_in[4];
  const float* b1  = (const float*)d_in[5];
  const float* w2l = (const float*)d_in[6];
  const float* w2r = (const float*)d_in[7];
  const float* b2  = (const float*)d_in[8];
  const float* wfc = (const float*)d_in[9];
  const float* bfc = (const float*)d_in[10];
  const float* fw0 = (const float*)d_in[11];
  const float* fb0 = (const float*)d_in[12];
  const float* fw1 = (const float*)d_in[13];
  const float* fb1 = (const float*)d_in[14];
  const float* fw2 = (const float*)d_in[15];
  const float* fb2 = (const float*)d_in[16];
  const float* fw3 = (const float*)d_in[17];
  const float* fb3 = (const float*)d_in[18];
  const float* fwo = (const float*)d_in[19];
  const float* fbo = (const float*)d_in[20];

  const int E = in_sizes[2] / 2;
  const int N = in_sizes[1] / C_N_FEAT;
  const int* src = ei;
  const int* dst = ei + E;

  // byte allocator, 1 KB aligned, +256B tail pad (GEMM prefetch overread safety)
  char* base = (char*)d_ws;
  size_t off = 0;
  auto alloc = [&](size_t bytes) {
    void* p = base + off;
    off += (bytes + 256 + 1023) & ~(size_t)1023;
    return p;
  };
  float*  cnt    = (float*)alloc((size_t)N * 4);
  float*  mean1  = (float*)alloc((size_t)N * 32 * 4);
  float*  mean2  = (float*)alloc((size_t)N * 128 * 4);
  float*  h1f    = (float*)alloc((size_t)N * 128 * 4);
  float*  h2f    = (float*)alloc((size_t)N * 64 * 4);
  float*  gmax   = (float*)alloc(64 * 4);
  float*  ctx    = (float*)alloc(C_CTX * 4);
  float*  z      = (float*)alloc((size_t)C_BT * C_D * 4);
  float*  logdet = (float*)alloc(C_BT * 4);
  float*  pbuf   = (float*)alloc((size_t)C_BT * C_OUTW * 4);
  __bf16* xb     = (__bf16*)alloc((size_t)N * 32 * 2);
  __bf16* mean1b = (__bf16*)alloc((size_t)N * 32 * 2);
  __bf16* h1b    = (__bf16*)alloc((size_t)N * 128 * 2);
  __bf16* mean2b = (__bf16*)alloc((size_t)N * 128 * 2);
  __bf16* hdin   = (__bf16*)alloc((size_t)C_BT * C_IN0P * 2);
  __bf16* fA     = (__bf16*)alloc((size_t)C_BT * C_HID * 2);
  __bf16* fB     = (__bf16*)alloc((size_t)C_BT * C_HID * 2);
  // transposed, masked bf16 weights
  __bf16* w1lT = (__bf16*)alloc((size_t)128 * 32 * 2);
  __bf16* w1rT = (__bf16*)alloc((size_t)128 * 32 * 2);
  __bf16* w2lT = (__bf16*)alloc((size_t)64 * 128 * 2);
  __bf16* w2rT = (__bf16*)alloc((size_t)64 * 128 * 2);
  __bf16* w0T  = (__bf16*)alloc((size_t)C_T * C_HID * C_IN0P * 2);
  __bf16* wh1T = (__bf16*)alloc((size_t)C_T * C_HID * C_HID * 2);
  __bf16* wh2T = (__bf16*)alloc((size_t)C_T * C_HID * C_HID * 2);
  __bf16* wh3T = (__bf16*)alloc((size_t)C_T * C_HID * C_HID * 2);
  __bf16* woT  = (__bf16*)alloc((size_t)C_T * C_OUTP * C_HID * 2);

  hipMemsetAsync(cnt,    0, (size_t)N * 4, stream);
  hipMemsetAsync(mean1,  0, (size_t)N * 32 * 4, stream);
  hipMemsetAsync(mean2,  0, (size_t)N * 128 * 4, stream);
  hipMemsetAsync(logdet, 0, (size_t)C_BT * 4, stream);

  // ---- weight prep (mask+transpose+bf16) ----
  auto prep = [&](const float* W, __bf16* Wt, int K, int Np, int Kpad, int Npad,
                  int mode, int parity) {
    int total = Npad * Kpad;
    prep_wt<<<(total + 255) / 256, 256, 0, stream>>>(W, Wt, K, Np, Kpad, Npad, mode, parity);
  };
  prep(w1l, w1lT, 32, 128, 32, 128, 0, 0);
  prep(w1r, w1rT, 32, 128, 32, 128, 0, 0);
  prep(w2l, w2lT, 128, 64, 128, 64, 0, 0);
  prep(w2r, w2rT, 128, 64, 128, 64, 0, 0);
  for (int t = 0; t < C_T; ++t) {
    int par = t & 1;
    prep(fw0 + (size_t)t * C_IN0 * C_HID, w0T + (size_t)t * C_HID * C_IN0P,
         C_IN0, C_HID, C_IN0P, C_HID, 1, par);
    prep(fw1 + (size_t)t * C_HID * C_HID, wh1T + (size_t)t * C_HID * C_HID,
         C_HID, C_HID, C_HID, C_HID, 2, par);
    prep(fw2 + (size_t)t * C_HID * C_HID, wh2T + (size_t)t * C_HID * C_HID,
         C_HID, C_HID, C_HID, C_HID, 2, par);
    prep(fw3 + (size_t)t * C_HID * C_HID, wh3T + (size_t)t * C_HID * C_HID,
         C_HID, C_HID, C_HID, C_HID, 2, par);
    prep(fwo + (size_t)t * C_HID * C_OUTW, woT + (size_t)t * C_OUTP * C_HID,
         C_HID, C_OUTW, C_HID, C_OUTP, 3, par);
  }

  // ---- SAGE layer 1 ----
  {
    long long n = (long long)N * 32;
    f32_to_bf16<<<(unsigned)((n + 255) / 256), 256, 0, stream>>>(x, xb, n);
  }
  {
    long long total = (long long)E * 4;   // F=32, 8 feats/thread
    scatter_accum<<<(unsigned)((total + 255) / 256), 256, 0, stream>>>(xb, src, dst, mean1, cnt, E, 32);
  }
  {
    long long total = (long long)N * 32;
    div_cnt_bf16<<<(unsigned)((total + 255) / 256), 256, 0, stream>>>(mean1, cnt, mean1b, total, 32);
  }
  {
    dim3 g((N + 31) / 32, 128 / 64);
    gemm_bf16<<<g, 128, 0, stream>>>(mean1b, 32, xb, 32, w1lT, w1rT, 32, b1,
                                     h1f, nullptr, 128, N, 128, 32, 32, 0);
  }
  rownorm_elu<<<N, 128, 0, stream>>>(h1f, h1b, nullptr, 128);

  // ---- SAGE layer 2 ----
  {
    long long total = (long long)E * 16;  // F=128
    scatter_accum<<<(unsigned)((total + 255) / 256), 256, 0, stream>>>(h1b, src, dst, mean2, nullptr, E, 128);
  }
  {
    long long total = (long long)N * 128;
    div_cnt_bf16<<<(unsigned)((total + 255) / 256), 256, 0, stream>>>(mean2, cnt, mean2b, total, 128);
  }
  {
    dim3 g((N + 31) / 32, 64 / 64);
    gemm_bf16<<<g, 128, 0, stream>>>(mean2b, 128, h1b, 128, w2lT, w2rT, 128, b2,
                                     h2f, nullptr, 64, N, 64, 128, 128, 0);
  }
  rownorm_elu<<<N, 64, 0, stream>>>(h2f, nullptr, h2f, 64);

  // ---- pool + ctx ----
  init_gmax<<<1, 64, 0, stream>>>(gmax);
  colmax<<<256, 64, 0, stream>>>(h2f, gmax, N, 64);
  ctx_proj<<<1, 32, 0, stream>>>(gmax, wfc, bfc, ctx);

  // ---- flow ----
  hipMemcpyAsync(z, theta, (size_t)C_BT * C_D * 4, hipMemcpyDeviceToDevice, stream);
  const dim3 gflowH(C_BT / 32, C_HID / 64);   // 128 x 4
  const dim3 gflowO(C_BT / 32, C_OUTP / 64);  // 128 x 6
  for (int t = 0; t < C_T; ++t) {
    pack_hdin<<<(C_BT * C_IN0P + 255) / 256, 256, 0, stream>>>(z, ctx, hdin);
    gemm_bf16<<<gflowH, 128, 0, stream>>>(hdin, C_IN0P, nullptr, 0,
                                          w0T + (size_t)t * C_HID * C_IN0P, nullptr, C_IN0P,
                                          fb0 + (size_t)t * C_HID, nullptr, fA, C_HID,
                                          C_BT, C_HID, C_IN0P, 0, 1);
    gemm_bf16<<<gflowH, 128, 0, stream>>>(fA, C_HID, nullptr, 0,
                                          wh1T + (size_t)t * C_HID * C_HID, nullptr, C_HID,
                                          fb1 + (size_t)t * C_HID, nullptr, fB, C_HID,
                                          C_BT, C_HID, C_HID, 0, 1);
    gemm_bf16<<<gflowH, 128, 0, stream>>>(fB, C_HID, nullptr, 0,
                                          wh2T + (size_t)t * C_HID * C_HID, nullptr, C_HID,
                                          fb2 + (size_t)t * C_HID, nullptr, fA, C_HID,
                                          C_BT, C_HID, C_HID, 0, 1);
    gemm_bf16<<<gflowH, 128, 0, stream>>>(fA, C_HID, nullptr, 0,
                                          wh3T + (size_t)t * C_HID * C_HID, nullptr, C_HID,
                                          fb3 + (size_t)t * C_HID, nullptr, fB, C_HID,
                                          C_BT, C_HID, C_HID, 0, 1);
    gemm_bf16<<<gflowO, 128, 0, stream>>>(fB, C_HID, nullptr, 0,
                                          woT + (size_t)t * C_OUTP * C_HID, nullptr, C_HID,
                                          fbo + (size_t)t * C_OUTW, pbuf, nullptr, C_OUTW,
                                          C_BT, C_OUTW, C_HID, 0, 0);
    rqs_kernel<<<C_BT / 256, 256, 0, stream>>>(pbuf, z, logdet);
  }
  logp_kernel<<<C_BT / 256, 256, 0, stream>>>(z, logdet, (float*)d_out);
}
